// MultiheadAttention_34059090657887
// MI455X (gfx1250) — matrix operand order repaired and checked by reading, not compiled, not run on hardware
//
#include <hip/hip_runtime.h>
#include <math.h>

typedef __bf16 bf16;
typedef __attribute__((ext_vector_type(16))) __bf16 v16bf;
typedef __attribute__((ext_vector_type(8)))  __bf16 v8bf;
typedef __attribute__((ext_vector_type(8)))  float  v8f;
typedef __attribute__((ext_vector_type(4)))  float  v4f;

#define BATCH 2
#define SEQ   2048
#define HID   1024
#define NHEAD 16
#define HDIM  64
#define MROWS (BATCH*SEQ)        // 4096
#define BN    (BATCH*NHEAD)      // 32

// ---------------------------------------------------------------------------
// WMMA fragment loaders (bf16, wave32, 16x16x32)
// A (16xK=32): lane l -> row (l&15); halves 0-7 <- K = hi*8+j ; halves 8-15 <- K = 16+hi*8+j
// B (K=32x16): lane l -> col (l&15); halves 0-7 <- K = hi*8+j ; halves 8-15 <- K = 16+hi*8+j
// ---------------------------------------------------------------------------
__device__ __forceinline__ v16bf load_a_frag(const bf16* base, int ld, int kb, int lane) {
    int r = lane & 15, hi = lane >> 4;
    const bf16* p = base + (size_t)r * ld + kb + hi * 8;
    v8bf lo = *(const v8bf*)p;
    v8bf h2 = *(const v8bf*)(p + 16);
    v16bf a;
#pragma unroll
    for (int i = 0; i < 8; i++) { a[i] = lo[i]; a[i + 8] = h2[i]; }
    return a;
}

__device__ __forceinline__ v16bf load_b_frag(const bf16* base, int ld, int kb, int lane) {
    int r = lane & 15, hi = lane >> 4;
    const bf16* p = base + (size_t)r * ld + kb + hi * 8;
    v8bf lo = *(const v8bf*)p;
    v8bf h2 = *(const v8bf*)(p + 16);
    v16bf b;
#pragma unroll
    for (int i = 0; i < 8; i++) { b[i] = lo[i]; b[i + 8] = h2[i]; }
    return b;
}

// A fragment sourced from f32 LDS (softmax probabilities), converted to bf16
__device__ __forceinline__ v16bf load_a_frag_f32(const float* base, int ld, int kb, int lane) {
    int r = lane & 15, hi = lane >> 4;
    const float* p = base + (size_t)r * ld + kb + hi * 8;
    v4f a0 = *(const v4f*)p;
    v4f a1 = *(const v4f*)(p + 4);
    v4f b0 = *(const v4f*)(p + 16);
    v4f b1 = *(const v4f*)(p + 20);
    v16bf o;
#pragma unroll
    for (int i = 0; i < 4; i++) {
        o[i]      = (bf16)a0[i];
        o[i + 4]  = (bf16)a1[i];
        o[i + 8]  = (bf16)b0[i];
        o[i + 12] = (bf16)b1[i];
    }
    return o;
}

__device__ __forceinline__ v8f wmma_bf16(v16bf a, v16bf b, v8f c) {
    return __builtin_amdgcn_wmma_f32_16x16x32_bf16(false, a, false, b, (short)0, c, false, false);
}

// ---------------------------------------------------------------------------
// Prep kernels
// ---------------------------------------------------------------------------
// Vectorized f32 -> bf16 convert: 8 elements per thread (b128 in, b128 out)
__global__ void to_bf16_kernel(const float* __restrict__ in, bf16* __restrict__ out, int n8) {
    for (int i = blockIdx.x * 256 + threadIdx.x; i < n8; i += gridDim.x * 256) {
        const float* p = in + (size_t)i * 8;
        v4f a = *(const v4f*)p;
        v4f b = *(const v4f*)(p + 4);
        v8bf o;
#pragma unroll
        for (int j = 0; j < 4; j++) { o[j] = (bf16)a[j]; o[j + 4] = (bf16)b[j]; }
        *(v8bf*)(out + (size_t)i * 8) = o;
    }
}

// in: f32 [n][n] row-major; out: bf16 [n][n] = in transposed (out[c][r] = in[r][c])
__global__ void transpose_bf16_kernel(const float* __restrict__ in, bf16* __restrict__ out, int n) {
    __shared__ float tile[32][33];
    int bx = blockIdx.x * 32, by = blockIdx.y * 32;
    int tx = threadIdx.x, ty = threadIdx.y;   // block (32,8)
#pragma unroll
    for (int i = ty; i < 32; i += 8)
        tile[i][tx] = in[(size_t)(by + i) * n + bx + tx];
    __syncthreads();
#pragma unroll
    for (int i = ty; i < 32; i += 8)
        out[(size_t)(bx + i) * n + by + tx] = (bf16)tile[tx][i];
}

// ---------------------------------------------------------------------------
// GEMM: C[M=4096 x N=1024] = A(bf16, MxK row-major) * Bt(bf16, NxK row-major)^T + bias
// block = 128 (4 waves); WG tile = 64 rows x 64 cols; wave tile = 64x16
// (4 M-subtiles share one B fragment -> 4 independent WMMAs per k-chunk)
// MODE 0: bf16 out, head layout   [b][nh][s][d]
// MODE 1: bf16 out, head-T layout [b][nh][d][s]
// MODE 2: f32  out, row-major     [m][n]
// ---------------------------------------------------------------------------
template <int MODE>
__global__ void gemm16_kernel(const bf16* __restrict__ A, const bf16* __restrict__ Bt,
                              const float* __restrict__ bias, void* __restrict__ out) {
    int wave = threadIdx.x >> 5, lane = threadIdx.x & 31;
    int row0 = blockIdx.y * 64;
    int col0 = blockIdx.x * 64 + wave * 16;
    const bf16* b_base = Bt + (size_t)col0 * HID;
    v8f acc[4] = {};
    for (int kb = 0; kb < HID; kb += 32) {
        v16bf b = load_b_frag(b_base, HID, kb, lane);
        v16bf a0 = load_a_frag(A + (size_t)(row0 +  0) * HID, HID, kb, lane);
        v16bf a1 = load_a_frag(A + (size_t)(row0 + 16) * HID, HID, kb, lane);
        v16bf a2 = load_a_frag(A + (size_t)(row0 + 32) * HID, HID, kb, lane);
        v16bf a3 = load_a_frag(A + (size_t)(row0 + 48) * HID, HID, kb, lane);
        acc[0] = wmma_bf16(a0, b, acc[0]);
        acc[1] = wmma_bf16(a1, b, acc[1]);
        acc[2] = wmma_bf16(a2, b, acc[2]);
        acc[3] = wmma_bf16(a3, b, acc[3]);
    }
    int col = col0 + (lane & 15);
    float bv = bias[col];
    int mr0 = (lane >> 4) * 8;
#pragma unroll
    for (int mt = 0; mt < 4; mt++) {
#pragma unroll
        for (int rr = 0; rr < 8; rr++) {
            int mrow = row0 + mt * 16 + mr0 + rr;
            float v = acc[mt][rr] + bv;
            if (MODE == 2) {
                ((float*)out)[(size_t)mrow * HID + col] = v;
            } else {
                int b_ = mrow >> 11, s = mrow & (SEQ - 1);
                int nh = col >> 6, d = col & (HDIM - 1);
                size_t idx = (MODE == 0)
                    ? ((((size_t)b_ * NHEAD + nh) * SEQ + s) * HDIM + d)
                    : ((((size_t)b_ * NHEAD + nh) * HDIM + d) * SEQ + s);
                ((bf16*)out)[idx] = (bf16)v;
            }
        }
    }
}

// ---------------------------------------------------------------------------
// Fused attention: per (bn, 16-query-row tile):
//   scores = Q K^T / 8  (WMMA, into LDS f32 strip 16x2048)
//   row softmax (LDS reductions), write normalized attn to global (coalesced)
//   ctx = P V (split-K WMMA, P from LDS, V^T from global), bf16 ctx out
// dynamic LDS = 16*2048*4 + 256*4 + 16*4 + 16*4 + 1024*4 = 136320 B
// ---------------------------------------------------------------------------
__global__ void attn_kernel(const bf16* __restrict__ Qh, const bf16* __restrict__ Kh,
                            const bf16* __restrict__ Vt, float* __restrict__ attn_out,
                            bf16* __restrict__ ctxB) {
    extern __shared__ char smem[];
    float* sc     = (float*)smem;                     // [16][2048]
    float* red    = (float*)(smem + 16 * SEQ * 4);    // [16][16]
    float* rowmax = red + 256;                        // [16]
    float* rowsum = rowmax + 16;                      // [16] (holds 1/sum)
    float* cred   = rowsum + 16;                      // [4][8][32]

    int bn = blockIdx.x;     // 0..31 (b*16+nh)
    int qt = blockIdx.y;     // 0..127
    int tid = threadIdx.x, wave = tid >> 5, lane = tid & 31;

    const bf16* Qbase = Qh + ((size_t)bn * SEQ + qt * 16) * HDIM;
    const bf16* Kbase = Kh + (size_t)bn * SEQ * HDIM;

    // Q tile fragments (K = 0..31, 32..63), shared by all waves
    v16bf qa0 = load_a_frag(Qbase, HDIM, 0, lane);
    v16bf qa1 = load_a_frag(Qbase, HDIM, 32, lane);

    // ---- scores: 128 column tiles of 16, 16 per wave ----
    for (int t = wave; t < SEQ / 16; t += 8) {
        const bf16* kb = Kbase + (size_t)t * 16 * HDIM;
        if (t + 8 < SEQ / 16)   // prefetch next K tile for this wave (2 KB)
            __builtin_prefetch((const char*)(kb + 8 * 16 * HDIM) + lane * 64, 0, 1);
        v16bf b0 = load_b_frag(kb, HDIM, 0, lane);
        v16bf b1 = load_b_frag(kb, HDIM, 32, lane);
        v8f acc = {};
        acc = wmma_bf16(qa0, b0, acc);
        acc = wmma_bf16(qa1, b1, acc);
        int n0 = t * 16 + (lane & 15);
        int mr0 = (lane >> 4) * 8;
#pragma unroll
        for (int r = 0; r < 8; r++)
            sc[(size_t)(mr0 + r) * SEQ + n0] = acc[r] * 0.125f;  // 1/sqrt(64)
    }
    __syncthreads();

    // ---- softmax over rows: 16 threads per row ----
    int r = tid >> 4, c0 = tid & 15;
    float m = -__builtin_inff();
    for (int c = c0; c < SEQ; c += 16) m = fmaxf(m, sc[(size_t)r * SEQ + c]);
    red[r * 16 + c0] = m;
    __syncthreads();
    if (tid < 16) {
        float mm = -__builtin_inff();
#pragma unroll
        for (int j = 0; j < 16; j++) mm = fmaxf(mm, red[tid * 16 + j]);
        rowmax[tid] = mm;
    }
    __syncthreads();
    float rm = rowmax[r];
    float s = 0.f;
    for (int c = c0; c < SEQ; c += 16) {
        float e = __expf(sc[(size_t)r * SEQ + c] - rm);
        sc[(size_t)r * SEQ + c] = e;
        s += e;
    }
    red[r * 16 + c0] = s;
    __syncthreads();
    if (tid < 16) {
        float ss = 0.f;
#pragma unroll
        for (int j = 0; j < 16; j++) ss += red[tid * 16 + j];
        rowsum[tid] = 1.0f / ss;
    }
    __syncthreads();

    // ---- normalize in LDS + coalesced attn store ----
    float* aout = attn_out + ((size_t)bn * SEQ + qt * 16) * SEQ;
    for (int i = tid; i < 16 * SEQ; i += 256) {
        int rr = i >> 11;
        float v = sc[i] * rowsum[rr];
        sc[i] = v;
        aout[i] = v;
    }
    __syncthreads();

    // ---- ctx = P V : wave w -> dtile = w&3, khalf = w>>2 (split-K x2) ----
    int dt = wave & 3, kh = wave >> 2;
    const bf16* vb = Vt + ((size_t)bn * HDIM + dt * 16) * SEQ;
    v8f acc = {};
    for (int kb = kh * (SEQ / 2); kb < (kh + 1) * (SEQ / 2); kb += 32) {
        if (kb + 32 < (kh + 1) * (SEQ / 2))  // prefetch next V^T chunk rows
            __builtin_prefetch((const char*)(vb + (size_t)(lane & 15) * SEQ + kb + 32) + (lane >> 4) * 32, 0, 1);
        v16bf a = load_a_frag_f32(sc, SEQ, kb, lane);
        v16bf b = load_b_frag(vb, SEQ, kb, lane);
        acc = wmma_bf16(a, b, acc);
    }
    if (kh == 1) {
#pragma unroll
        for (int rr = 0; rr < 8; rr++) cred[(dt * 8 + rr) * 32 + lane] = acc[rr];
    }
    __syncthreads();
    if (kh == 0) {
        int b_ = bn >> 4, nh = bn & 15;
        int col = nh * HDIM + dt * 16 + (lane & 15);
        int mr0 = (lane >> 4) * 8;
#pragma unroll
        for (int rr = 0; rr < 8; rr++) {
            float v = acc[rr] + cred[(dt * 8 + rr) * 32 + lane];
            int srow = qt * 16 + mr0 + rr;
            ctxB[((size_t)b_ * SEQ + srow) * (NHEAD * HDIM) + col] = (bf16)v;
        }
    }
}

// ---------------------------------------------------------------------------
// Launch
// ---------------------------------------------------------------------------
extern "C" void kernel_launch(void* const* d_in, const int* in_sizes, int n_in,
                              void* d_out, int out_size, void* d_ws, size_t ws_size,
                              hipStream_t stream) {
    const float* query = (const float*)d_in[0];
    const float* key_  = (const float*)d_in[1];
    const float* value = (const float*)d_in[2];
    // d_in[3] attention_mask: no effect in reference math (faithfully ignored)
    const float* Wq = (const float*)d_in[4];
    const float* bq = (const float*)d_in[5];
    const float* Wk = (const float*)d_in[6];
    const float* bk = (const float*)d_in[7];
    const float* Wv = (const float*)d_in[8];
    const float* bv = (const float*)d_in[9];
    const float* Wo = (const float*)d_in[10];
    const float* bo = (const float*)d_in[11];

    // workspace carve (bytes)
    const size_t MB = 1024 * 1024;
    char* ws = (char*)d_ws;
    bf16* XqB = (bf16*)(ws + 0 * MB);    // 8 MB each activation
    bf16* XkB = (bf16*)(ws + 8 * MB);
    bf16* XvB = (bf16*)(ws + 16 * MB);
    bf16* WqT = (bf16*)(ws + 24 * MB);   // 2 MB each weight
    bf16* WkT = (bf16*)(ws + 26 * MB);
    bf16* WvT = (bf16*)(ws + 28 * MB);
    bf16* WoT = (bf16*)(ws + 30 * MB);
    bf16* Qh  = (bf16*)(ws + 32 * MB);   // 8 MB each
    bf16* Kh  = (bf16*)(ws + 40 * MB);
    bf16* Vt  = (bf16*)(ws + 48 * MB);
    bf16* ctx = (bf16*)(ws + 56 * MB);

    float* out_main = (float*)d_out;                       // [4096,1024]
    float* out_attn = (float*)d_out + (size_t)MROWS * HID; // [32,2048,2048]

    const int NACT8 = MROWS * HID / 8;
    to_bf16_kernel<<<2048, 256, 0, stream>>>(query, XqB, NACT8);
    to_bf16_kernel<<<2048, 256, 0, stream>>>(key_,  XkB, NACT8);
    to_bf16_kernel<<<2048, 256, 0, stream>>>(value, XvB, NACT8);

    dim3 tb(32, 8), tg(HID / 32, HID / 32);
    transpose_bf16_kernel<<<tg, tb, 0, stream>>>(Wq, WqT, HID);
    transpose_bf16_kernel<<<tg, tb, 0, stream>>>(Wk, WkT, HID);
    transpose_bf16_kernel<<<tg, tb, 0, stream>>>(Wv, WvT, HID);
    transpose_bf16_kernel<<<tg, tb, 0, stream>>>(Wo, WoT, HID);

    dim3 gg(HID / 64, MROWS / 64);  // (16, 64)
    gemm16_kernel<0><<<gg, 128, 0, stream>>>(XqB, WqT, bq, (void*)Qh);
    gemm16_kernel<0><<<gg, 128, 0, stream>>>(XkB, WkT, bk, (void*)Kh);
    gemm16_kernel<1><<<gg, 128, 0, stream>>>(XvB, WvT, bv, (void*)Vt);

    const int ATT_SMEM = 16 * SEQ * 4 + 256 * 4 + 16 * 4 + 16 * 4 + 1024 * 4; // 136320 B
    (void)hipFuncSetAttribute((const void*)attn_kernel,
                              hipFuncAttributeMaxDynamicSharedMemorySize, ATT_SMEM);
    attn_kernel<<<dim3(BN, SEQ / 16), 256, ATT_SMEM, stream>>>(Qh, Kh, Vt, out_attn, ctx);

    gemm16_kernel<2><<<gg, 128, 0, stream>>>(ctx, WoT, bo, (void*)out_main);
}
